// SE3Transformer_6966436954872
// MI455X (gfx1250) — matrix operand order repaired and checked by reading, not compiled, not run on hardware
//
#include <hip/hip_runtime.h>
#include <math.h>

#define N_NODES 16384
#define DEG 20
#define MUL 8
#define NB 10
#define HID 64
#define RMAXF 3.5f

#define BLK_NODES 8
#define BLK_EDGES 160      // BLK_NODES * DEG
#define BLK_THREADS 320    // 10 wave32s, 16 edges per wave

typedef __attribute__((ext_vector_type(16))) _Float16 v16h;
typedef __attribute__((ext_vector_type(8)))  float    v8f;

__device__ __forceinline__ v8f wmma_f16(v16h a, v16h b, v8f c) {
  // D = A(16x32 f16) * B(32x16 f16) + C(16x16 f32)
  return __builtin_amdgcn_wmma_f32_16x16x32_f16(false, a, false, b, (short)0, c,
                                                false, false);
}

// single-instruction v_rcp_f32 instead of IEEE divide expansion
__device__ __forceinline__ float fast_rcp(float x) { return __builtin_amdgcn_rcpf(x); }

// ---------------------------------------------------------------------------
// Kernel 1: per-node query + pre-contraction with Wd.
// qW[n][0..7]      = sum_i qs[i]*Wd_s[i][j]
// qW[n][8+j*3+c]   = sum_i qv[i][c]*Wd_v[i][j]
// ---------------------------------------------------------------------------
__global__ void prep_nodes(const float* __restrict__ f,
                           const float* __restrict__ Wq_s,
                           const float* __restrict__ Wq_v,
                           const float* __restrict__ Wd_s,
                           const float* __restrict__ Wd_v,
                           float* __restrict__ qW) {
  int n = blockIdx.x * blockDim.x + threadIdx.x;
  if (n >= N_NODES) return;
  const float* fr = f + (size_t)n * 32;
  float fs[MUL], fv[MUL][3];
#pragma unroll
  for (int i = 0; i < MUL; ++i) fs[i] = fr[i];
#pragma unroll
  for (int i = 0; i < MUL; ++i)
#pragma unroll
    for (int c = 0; c < 3; ++c) fv[i][c] = fr[8 + i * 3 + c];

  const float inv_sqrt_mul = 0.3535533906f; // 1/sqrt(8)
  float qs[MUL], qv[MUL][3];
#pragma unroll
  for (int o = 0; o < MUL; ++o) {
    float a = 0.f, vx = 0.f, vy = 0.f, vz = 0.f;
#pragma unroll
    for (int i = 0; i < MUL; ++i) {
      float w = Wq_s[i * MUL + o];
      a += fs[i] * w;
      float wv = Wq_v[i * MUL + o];
      vx += fv[i][0] * wv; vy += fv[i][1] * wv; vz += fv[i][2] * wv;
    }
    qs[o] = a * inv_sqrt_mul;
    qv[o][0] = vx * inv_sqrt_mul; qv[o][1] = vy * inv_sqrt_mul; qv[o][2] = vz * inv_sqrt_mul;
  }
  float outr[32];
#pragma unroll
  for (int j = 0; j < MUL; ++j) {
    float a = 0.f, vx = 0.f, vy = 0.f, vz = 0.f;
#pragma unroll
    for (int i = 0; i < MUL; ++i) {
      a  += qs[i] * Wd_s[i * MUL + j];
      float w = Wd_v[i * MUL + j];
      vx += qv[i][0] * w; vy += qv[i][1] * w; vz += qv[i][2] * w;
    }
    outr[j] = a;
    outr[8 + j * 3 + 0] = vx; outr[8 + j * 3 + 1] = vy; outr[8 + j * 3 + 2] = vz;
  }
  float* dst = qW + (size_t)n * 32;
#pragma unroll
  for (int j = 0; j < 32; ++j) dst[j] = outr[j];
}

// ---------------------------------------------------------------------------
// Kernel 2: repack Wk2/Wv2 (64x256 f32) into f16 B-fragment order.
// Fragment = 32x16 slice of B for WMMA_F32_16X16X32_F16:
//   lane n (0..15):  N=n, K = frag*32 + h        (h=0..15)
//   lane n+16:       N=n, K = frag*32 + 16 + h
// Folds 1/sqrt(HID)=1/8 normalization.
// Linear: pw2[(((m*16 + t)*2 + frag)*32 + lane)*16 + h]
// ---------------------------------------------------------------------------
__global__ void pack_w2(const float* __restrict__ Wk2,
                        const float* __restrict__ Wv2,
                        _Float16* __restrict__ pw2) {
  int idx = blockIdx.x * blockDim.x + threadIdx.x;
  if (idx >= 2 * 16 * 2 * 32 * 16) return;
  int h    = idx & 15;
  int lane = (idx >> 4) & 31;
  int rest = idx >> 9;          // frag + 2*(t + 16*m)
  int frag = rest & 1;
  int t    = (rest >> 1) & 15;
  int m    = rest >> 5;
  int n    = lane & 15;
  int khi  = lane >> 4;
  int k    = 32 * frag + 16 * khi + h;
  int col  = 16 * t + n;
  const float* src = m ? Wv2 : Wk2;
  pw2[idx] = (_Float16)(src[k * 256 + col] * 0.125f);
}

// ---------------------------------------------------------------------------
// Kernel 3: fused edge kernel. One block = 8 dst nodes = 160 edges = 10 waves.
// ---------------------------------------------------------------------------
__global__ __launch_bounds__(BLK_THREADS)
void edge_attn(const float* __restrict__ f,
               const float* __restrict__ pos,
               const float* __restrict__ Wk1,
               const float* __restrict__ Wv1,
               const int* __restrict__ edge_src,
               const int* __restrict__ edge_dst,
               const float* __restrict__ qW,
               const _Float16* __restrict__ pw2,
               float* __restrict__ out) {
  __shared__ float sXs  [BLK_EDGES * 8];    // scalars (gathered from src)
  __shared__ float sXv  [BLK_EDGES * 24];   // vectors
  __shared__ float sB   [BLK_EDGES * 8];    // dot(xv_i, unit) = dot(xv_i,s1)/sqrt3
  __shared__ float sEmb [BLK_EDGES * 10];   // radial basis
  __shared__ float sGeom[BLK_EDGES * 4];    // s1x,s1y,s1z,cutoff
  __shared__ float sA2  [BLK_EDGES * 8];    // sum_c unit_c * qWdv[dst][o][c]
  __shared__ float sQd  [BLK_NODES * 32];   // qWds(8) + qWdv(24) per dst node
  __shared__ float sLogit[BLK_EDGES];
  __shared__ float sCoef [BLK_EDGES];
  __shared__ float sVout [BLK_EDGES * 32];  // per-edge message (vs 8, vv 24)

  const int tid = threadIdx.x;
  const int n0  = blockIdx.x * BLK_NODES;
  const int e0  = n0 * DEG;

  // ---------------- per-edge prep -----------------
  float u0 = 0.f, u1 = 0.f, u2 = 0.f;
  if (tid < BLK_EDGES) {
    int eg  = e0 + tid;
    int src = edge_src[eg];
    int dstn = edge_dst[eg];
    float xr[32];
    const float4* frow = reinterpret_cast<const float4*>(f + (size_t)src * 32);
#pragma unroll
    for (int q = 0; q < 8; ++q) {
      float4 v = frow[q];
      xr[4 * q + 0] = v.x; xr[4 * q + 1] = v.y; xr[4 * q + 2] = v.z; xr[4 * q + 3] = v.w;
    }
    float px = pos[src * 3 + 0] - pos[dstn * 3 + 0];
    float py = pos[src * 3 + 1] - pos[dstn * 3 + 1];
    float pz = pos[src * 3 + 2] - pos[dstn * 3 + 2];
    float r = sqrtf(px * px + py * py + pz * pz);
    float inv_r = fast_rcp(r);
    u0 = px * inv_r; u1 = py * inv_r; u2 = pz * inv_r;
    // soft_one_hot_linspace, basis='smooth_finite', cutoff=True, * sqrt(NB)
    const float inv_step = 11.f / RMAXF;
#pragma unroll
    for (int k = 0; k < NB; ++k) {
      float center = RMAXF * (float)(k + 1) / 11.f;
      float d  = (r - center) * inv_step;
      float a1 = d + 1.f, a2 = 1.f - d;
      float b1 = (a1 > 0.f) ? __expf(-fast_rcp(a1)) : 0.f;
      float b2 = (a2 > 0.f) ? __expf(-fast_rcp(a2)) : 0.f;
      sEmb[tid * 10 + k] = 26.66909f * b1 * b2;  // 1.14136*e^2*sqrt(10)
    }
    float cx = 10.f * (1.f - r * (1.f / RMAXF));
    float cutoff = (cx > 0.f) ? __expf(-fast_rcp(cx)) : 0.f;
    const float s3 = 1.7320508f;
    sGeom[tid * 4 + 0] = s3 * u0;
    sGeom[tid * 4 + 1] = s3 * u1;
    sGeom[tid * 4 + 2] = s3 * u2;
    sGeom[tid * 4 + 3] = cutoff;
#pragma unroll
    for (int i = 0; i < MUL; ++i) sXs[tid * 8 + i] = xr[i];
#pragma unroll
    for (int i = 0; i < MUL; ++i) {
      float vx = xr[8 + i * 3 + 0], vy = xr[8 + i * 3 + 1], vz = xr[8 + i * 3 + 2];
      sXv[tid * 24 + i * 3 + 0] = vx;
      sXv[tid * 24 + i * 3 + 1] = vy;
      sXv[tid * 24 + i * 3 + 2] = vz;
      sB[tid * 8 + i] = vx * u0 + vy * u1 + vz * u2;
    }
  } else {
    for (int j = tid - BLK_EDGES; j < BLK_NODES * 32; j += BLK_THREADS - BLK_EDGES)
      sQd[j] = qW[(size_t)n0 * 32 + j];
  }
  __syncthreads();
  if (tid < BLK_EDGES) {
    int dl = tid / DEG;
#pragma unroll
    for (int o = 0; o < MUL; ++o)
      sA2[tid * 8 + o] = u0 * sQd[dl * 32 + 8 + o * 3 + 0] +
                         u1 * sQd[dl * 32 + 8 + o * 3 + 1] +
                         u2 * sQd[dl * 32 + 8 + o * 3 + 2];
  }
  __syncthreads();

  // ---------------- WMMA phase: each wave owns 16 edges -----------------
  const int wv      = tid >> 5;
  const int lane    = tid & 31;
  const int mrow    = lane & 15;   // A-matrix row (edge within wave tile)
  const int khi     = lane >> 4;   // A K-interleave half
  const int ebase   = wv * 16;
  const int rowbase = khi * 8;     // C/D rows held by this lane
  const int o_lane  = lane & 7;    // output-channel o of this lane's column
  const int ihalf   = (lane >> 3) & 1;

  // per-lane row constants, hoisted out of the tile loops
  int eArr[8];
#pragma unroll
  for (int g = 0; g < 8; ++g) eArr[g] = ebase + rowbase + g;

  float eb[NB];
  {
    const int eA = ebase + mrow;
#pragma unroll
    for (int q = 0; q < NB; ++q) eb[q] = sEmb[eA * 10 + q];
  }

  // FC1 + silu straight into A-fragment slots (contiguous k = 16*khi + h, same order as pw2 B)
  auto buildA = [&](const float* __restrict__ W1, v16h& a0, v16h& a1) {
    const float inv_sqrt_nb = 0.316227766f; // 1/sqrt(10)
#pragma unroll
    for (int fb = 0; fb < 2; ++fb) {
#pragma unroll
      for (int h = 0; h < 16; ++h) {
        int klocal = 16 * khi + h;
        int k = 32 * fb + klocal;
        float acc = 0.f;
#pragma unroll
        for (int q = 0; q < NB; ++q) acc += eb[q] * W1[q * HID + k];
        acc *= inv_sqrt_nb;
        float s = acc * fast_rcp(1.f + __expf(-acc));  // silu, v_rcp_f32
        if (fb == 0) a0[h] = (_Float16)s; else a1[h] = (_Float16)s;
      }
    }
  };

  // ---- V pipeline: weights -> per-edge message (vs, vv) ----
  {
    v16h a0, a1;
    buildA(Wv1, a0, a1);
    float accS[8];
    float accV[8][3];
    float s1g[8][3];
#pragma unroll
    for (int g = 0; g < 8; ++g) {
      accS[g] = 0.f; accV[g][0] = 0.f; accV[g][1] = 0.f; accV[g][2] = 0.f;
      s1g[g][0] = sGeom[eArr[g] * 4 + 0];
      s1g[g][1] = sGeom[eArr[g] * 4 + 1];
      s1g[g][2] = sGeom[eArr[g] * 4 + 2];
    }
#pragma unroll
    for (int t = 0; t < 16; ++t) {
      const v16h b0 = *reinterpret_cast<const v16h*>(pw2 + (((16 + t) * 2 + 0) * 32 + lane) * 16);
      const v16h b1 = *reinterpret_cast<const v16h*>(pw2 + (((16 + t) * 2 + 1) * 32 + lane) * 16);
      v8f d = {};
      d = wmma_f16(a0, b0, d);
      d = wmma_f16(a1, b1, d);
      const int path = t >> 2;
      const int i    = ((t & 3) << 1) + ihalf;
#pragma unroll
      for (int g = 0; g < 8; ++g) {
        int e = eArr[g];
        float w = d[g];
        if (path == 0) {
          accS[g] += w * sXs[e * 8 + i];
        } else if (path == 1) {
          accS[g] += w * sB[e * 8 + i];
        } else if (path == 2) {
          float x = w * sXs[e * 8 + i];
          accV[g][0] += x * s1g[g][0];
          accV[g][1] += x * s1g[g][1];
          accV[g][2] += x * s1g[g][2];
        } else {
          accV[g][0] += w * sXv[e * 24 + i * 3 + 0];
          accV[g][1] += w * sXv[e * 24 + i * 3 + 1];
          accV[g][2] += w * sXv[e * 24 + i * 3 + 2];
        }
      }
    }
    // sum over the two i-halves (lanes n and n+8 share o)
#pragma unroll
    for (int g = 0; g < 8; ++g) {
      accS[g]    += __shfl_xor(accS[g], 8, 32);
      accV[g][0] += __shfl_xor(accV[g][0], 8, 32);
      accV[g][1] += __shfl_xor(accV[g][1], 8, 32);
      accV[g][2] += __shfl_xor(accV[g][2], 8, 32);
    }
    const float NORM = 0.25f; // 1/(sqrt(MUL)*sqrt(2))
    if (ihalf == 0) {
#pragma unroll
      for (int g = 0; g < 8; ++g) {
        int e = eArr[g];
        sVout[e * 32 + o_lane] = accS[g] * NORM;
        sVout[e * 32 + 8 + o_lane * 3 + 0] = accV[g][0] * NORM;
        sVout[e * 32 + 8 + o_lane * 3 + 1] = accV[g][1] * NORM;
        sVout[e * 32 + 8 + o_lane * 3 + 2] = accV[g][2] * NORM;
      }
    }
  }

  // ---- K pipeline: weights -> logits (linear functional of wk) ----
  {
    v16h a0, a1;
    buildA(Wk1, a0, a1);
    float lacc[8];
    float qds[8], qdv[8][3];
#pragma unroll
    for (int g = 0; g < 8; ++g) {
      lacc[g] = 0.f;
      int dl = eArr[g] / DEG;              // hoisted: constant per row slot
      qds[g]    = sQd[dl * 32 + o_lane];
      qdv[g][0] = sQd[dl * 32 + 8 + o_lane * 3 + 0];
      qdv[g][1] = sQd[dl * 32 + 8 + o_lane * 3 + 1];
      qdv[g][2] = sQd[dl * 32 + 8 + o_lane * 3 + 2];
    }
#pragma unroll
    for (int t = 0; t < 16; ++t) {
      const v16h b0 = *reinterpret_cast<const v16h*>(pw2 + ((t * 2 + 0) * 32 + lane) * 16);
      const v16h b1 = *reinterpret_cast<const v16h*>(pw2 + ((t * 2 + 1) * 32 + lane) * 16);
      v8f d = {};
      d = wmma_f16(a0, b0, d);
      d = wmma_f16(a1, b1, d);
      const int path = t >> 2;
      const int i    = ((t & 3) << 1) + ihalf;
#pragma unroll
      for (int g = 0; g < 8; ++g) {
        int e = eArr[g];
        float w = d[g];
        float coef;
        if (path == 0)      coef = sXs[e * 8 + i] * qds[g];
        else if (path == 1) coef = sB [e * 8 + i] * qds[g];
        else if (path == 2) coef = sXs[e * 8 + i] * sA2[e * 8 + o_lane];
        else coef = 0.57735027f *
              (sXv[e * 24 + i * 3 + 0] * qdv[g][0] +
               sXv[e * 24 + i * 3 + 1] * qdv[g][1] +
               sXv[e * 24 + i * 3 + 2] * qdv[g][2]);
        lacc[g] += w * coef;
      }
    }
    // sum over all 16 columns of the half-wave (o and i)
#pragma unroll
    for (int g = 0; g < 8; ++g) {
      lacc[g] += __shfl_xor(lacc[g], 1, 32);
      lacc[g] += __shfl_xor(lacc[g], 2, 32);
      lacc[g] += __shfl_xor(lacc[g], 4, 32);
      lacc[g] += __shfl_xor(lacc[g], 8, 32);
    }
    const float LOGIT_SCALE = 0.0220970869f; // 0.25 / (MUL*sqrt(2))
    if ((lane & 15) == 0) {
#pragma unroll
      for (int g = 0; g < 8; ++g)
        sLogit[eArr[g]] = lacc[g] * LOGIT_SCALE;
    }
  }
  __syncthreads();

  // ---------------- scatter softmax (20 consecutive edges per dst) ----------
  if (tid < BLK_NODES) {
    int base = tid * DEG;
    float mx = -1e30f;
#pragma unroll
    for (int k = 0; k < DEG; ++k) mx = fmaxf(mx, sLogit[base + k]);
    float ew[DEG];
    float z = 0.f;
#pragma unroll
    for (int k = 0; k < DEG; ++k) {
      float e_ = sGeom[(base + k) * 4 + 3] * __expf(sLogit[base + k] - mx);
      ew[k] = e_; z += e_;
    }
    if (z == 0.f) z = 1.f;
    float invz = fast_rcp(z);
#pragma unroll
    for (int k = 0; k < DEG; ++k)
      sCoef[base + k] = sqrtf(ew[k] * invz + 1e-12f);
  }
  __syncthreads();

  // ---------------- weighted segment-sum -> output ----------
  if (tid < BLK_NODES * 32) {
    int g = tid >> 5, comp = tid & 31;
    float acc = 0.f;
#pragma unroll
    for (int k = 0; k < DEG; ++k)
      acc += sCoef[g * DEG + k] * sVout[(g * DEG + k) * 32 + comp];
    out[(size_t)(n0 + g) * 32 + comp] = acc;
  }
}

// ---------------------------------------------------------------------------
extern "C" void kernel_launch(void* const* d_in, const int* in_sizes, int n_in,
                              void* d_out, int out_size, void* d_ws, size_t ws_size,
                              hipStream_t stream) {
  const float* f    = (const float*)d_in[0];
  const float* pos  = (const float*)d_in[1];
  const float* Wq_s = (const float*)d_in[2];
  const float* Wq_v = (const float*)d_in[3];
  const float* Wk1  = (const float*)d_in[4];
  const float* Wk2  = (const float*)d_in[5];
  const float* Wv1  = (const float*)d_in[6];
  const float* Wv2  = (const float*)d_in[7];
  const float* Wd_s = (const float*)d_in[8];
  const float* Wd_v = (const float*)d_in[9];
  const int* esrc   = (const int*)d_in[10];
  const int* edst   = (const int*)d_in[11];
  float* out = (float*)d_out;

  float*    qW  = (float*)d_ws;                                        // N*32 f32 = 2 MB
  _Float16* pw2 = (_Float16*)((char*)d_ws + (size_t)N_NODES * 32 * 4); // 64 KB f16

  prep_nodes<<<N_NODES / 256, 256, 0, stream>>>(f, Wq_s, Wq_v, Wd_s, Wd_v, qW);
  pack_w2<<<(2 * 16 * 2 * 32 * 16) / 256, 256, 0, stream>>>(Wk2, Wv2, pw2);
  edge_attn<<<N_NODES / BLK_NODES, BLK_THREADS, 0, stream>>>(
      f, pos, Wk1, Wv1, esrc, edst, qW, pw2, out);
}
